// STGAT_ActorCritic_15384572854699
// MI455X (gfx1250) — matrix operand order repaired and checked by reading, not compiled, not run on hardware
//
#include <hip/hip_runtime.h>
#include <math.h>

#define BATCH 8
#define NNODE 1024
#define NHEAD 3
#define HID   64
#define FDIM  (NHEAD*HID)   /* 192 */
#define LRA   0.2f
#define GM    (BATCH*NNODE) /* 8192 rows for GRU GEMM */

typedef _Float16 half_t;
typedef _Float16 v16h __attribute__((ext_vector_type(16)));
typedef _Float16 v8h  __attribute__((ext_vector_type(8)));
typedef float    v8f  __attribute__((ext_vector_type(8)));

// ---------------------------------------------------------------------------
// Wh = in @ W (per node) + per-node attention scalars s,d.
// One block per (b,n), 192 threads. Wh written f32 in (B,H,N,HID).
// ---------------------------------------------------------------------------
__global__ void wh_kernel(const float* __restrict__ in, int fin,
                          const float* __restrict__ W,
                          const float* __restrict__ asrc,
                          const float* __restrict__ adst,
                          float* __restrict__ wh32,
                          float* __restrict__ sbuf, float* __restrict__ dbuf) {
    int bn = blockIdx.x;
    int b  = bn / NNODE, n = bn % NNODE;
    int g  = threadIdx.x;                // 0..191
    __shared__ float xin[64];
    __shared__ float shW[FDIM];
    if (g < fin) xin[g] = in[(size_t)bn * fin + g];
    __syncthreads();
    int h = g / HID, f = g % HID;
    float acc = 0.f;
    for (int k = 0; k < fin; ++k) acc += xin[k] * W[k * FDIM + g];
    wh32[((size_t)(b * NHEAD + h) * NNODE + n) * HID + f] = acc;
    shW[g] = acc;
    __syncthreads();
    if (g < 2 * NHEAD) {
        int hh = g % NHEAD;
        const float* av = (g < NHEAD) ? asrc : adst;
        float s = 0.f;
        for (int k = 0; k < HID; ++k) s += shW[hh * HID + k] * av[hh * HID + k];
        float* dst = (g < NHEAD) ? sbuf : dbuf;
        dst[(size_t)(b * NHEAD + hh) * NNODE + n] = s;
    }
}

// ---------------------------------------------------------------------------
// (B,H,N,HID) f32 -> (B,H,HID,N) f16 (WMMA B-operand layout: contiguous K).
// One block per 64x64 tile; LDS transpose, coalesced in and out.
// ---------------------------------------------------------------------------
__global__ void __launch_bounds__(256) transpose_kernel(
    const float* __restrict__ wh32, half_t* __restrict__ whT) {
    const int ntile = NNODE / 64;
    int bh = blockIdx.x / ntile;
    int n0 = (blockIdx.x % ntile) * 64;
    int tid = threadIdx.x;
    __shared__ float tile[64][65];
    const float* src = wh32 + ((size_t)bh * NNODE + n0) * HID;
#pragma unroll
    for (int k = 0; k < 16; ++k) {
        int idx = k * 256 + tid;
        int n = idx >> 6, f = idx & 63;
        tile[n][f] = src[(size_t)n * HID + f];
    }
    __syncthreads();
#pragma unroll
    for (int k = 0; k < 16; ++k) {
        int idx = k * 256 + tid;
        int f = idx >> 6, n = idx & 63;
        whT[((size_t)bh * HID + f) * NNODE + n0 + n] = (half_t)tile[n][f];
    }
}

// ---------------------------------------------------------------------------
// Per-row softmax stats: m_i, 1/sum_j exp(e_ij - m_i). One wave per row.
// ---------------------------------------------------------------------------
__global__ void stats_kernel(const float* __restrict__ sbuf,
                             const float* __restrict__ dbuf,
                             const int* __restrict__ adj,
                             float* __restrict__ mbuf,
                             float* __restrict__ invbuf) {
    int wave = threadIdx.x >> 5, lane = threadIdx.x & 31;
    int row = blockIdx.x * 8 + wave;     // index into (B*H*N)
    int bh = row / NNODE;
    int i  = row % NNODE;
    float sv = sbuf[row];
    const float* drow = dbuf + (size_t)bh * NNODE;
    const int*   arow = adj + (size_t)i * NNODE;
    float mx = -3.0e38f;
    for (int j = lane; j < NNODE; j += 32) {
        float v = sv + drow[j];
        float e = (arow[j] > 0) ? (v > 0.f ? v : LRA * v) : -1.0e9f;
        mx = fmaxf(mx, e);
    }
    for (int off = 16; off > 0; off >>= 1) mx = fmaxf(mx, __shfl_xor(mx, off));
    float sum = 0.f;
    for (int j = lane; j < NNODE; j += 32) {
        float v = sv + drow[j];
        float e = (arow[j] > 0) ? (v > 0.f ? v : LRA * v) : -1.0e9f;
        sum += __expf(e - mx);
    }
    for (int off = 16; off > 0; off >>= 1) sum += __shfl_xor(sum, off);
    if (lane == 0) { mbuf[row] = mx; invbuf[row] = 1.0f / sum; }
}

// ---------------------------------------------------------------------------
// Main kernel: hp[b,h,i,:] = softmax_j(e) @ Wh via f16 WMMA.
// Block = 4 waves = one 16(i) x 64(f) tile. K-chunk = 64 j's per iteration:
//  - P tile 16x64 regenerated straight-line (8 j's/thread, int4 adj loads,
//    one ds_store_b128 per thread)
//  - B fragments loaded DIRECTLY from global (pre-transposed f16 Wh)
//  - two v_wmma_f32_16x16x32_f16 per wave per barrier pair.
// ---------------------------------------------------------------------------
__global__ void __launch_bounds__(128) attn_kernel(
    const half_t* __restrict__ whT, const float* __restrict__ sbuf,
    const float* __restrict__ dbuf, const float* __restrict__ mbuf,
    const float* __restrict__ invbuf, const int* __restrict__ adj,
    float* __restrict__ hp) {
    const int ntiles = NNODE / 16;
    int itile = blockIdx.x % ntiles;
    int bh    = blockIdx.x / ntiles;
    int tid = threadIdx.x, wave = tid >> 5, lane = tid & 31;
    const int i0 = itile * 16;

    __shared__ __align__(32) half_t pA[16 * 64];    // P tile, row-major (i,k)
    __shared__ float srow[16], smax[16], sinv[16];

    if (tid < 16) {
        size_t r = (size_t)bh * NNODE + i0 + tid;
        srow[tid] = sbuf[r]; smax[tid] = mbuf[r]; sinv[tid] = invbuf[r];
    }

    v8f acc = {0.f, 0.f, 0.f, 0.f, 0.f, 0.f, 0.f, 0.f};
    const int pi  = tid >> 3;            // P row handled by this thread
    const int pj8 = (tid & 7) << 3;      // 8 consecutive j's
    const int mrow = lane & 15;
    const int klo  = (lane < 16) ? 0 : 16;
    const int kb   = (lane < 16) ? 0 : 16;
    const int col  = wave * 16 + (lane & 15);
    const half_t* bcol = whT + ((size_t)bh * HID + col) * NNODE;

    for (int j0 = 0; j0 < NNODE; j0 += 64) {
        __syncthreads();   // previous frag reads done; srow visible on iter 0
        // ---- regenerate P (16 x 64) f16, 8 elements per thread ----
        {
            const float* dptr = dbuf + (size_t)bh * NNODE + j0 + pj8;
            float4 d0 = *(const float4*)(dptr);
            float4 d1 = *(const float4*)(dptr + 4);
            const int* aptr = adj + (size_t)(i0 + pi) * NNODE + j0 + pj8;
            int4 a0 = *(const int4*)(aptr);
            int4 a1 = *(const int4*)(aptr + 4);
            float dd[8] = {d0.x, d0.y, d0.z, d0.w, d1.x, d1.y, d1.z, d1.w};
            int   aa[8] = {a0.x, a0.y, a0.z, a0.w, a1.x, a1.y, a1.z, a1.w};
            float sv = srow[pi], mv = smax[pi], iv = sinv[pi];
            union { v8h v; half_t h[8]; } P;
#pragma unroll
            for (int q = 0; q < 8; ++q) {
                float vv = sv + dd[q];
                float e = (aa[q] > 0) ? (vv > 0.f ? vv : LRA * vv) : -1.0e9f;
                P.h[q] = (half_t)(__expf(e - mv) * iv);
            }
            *reinterpret_cast<v8h*>(&pA[pi * 64 + pj8]) = P.v;
        }
        __syncthreads();   // pA visible
        // ---- two K=32 WMMA steps ----
#pragma unroll
        for (int c = 0; c < 2; ++c) {
            int off = c * 32;
            v8h alo = *reinterpret_cast<const v8h*>(&pA[mrow * 64 + off + klo]);
            v8h ahi = *reinterpret_cast<const v8h*>(&pA[mrow * 64 + off + klo + 8]);
            v16h afrag = __builtin_shufflevector(alo, ahi,
                           0, 1, 2, 3, 4, 5, 6, 7, 8, 9, 10, 11, 12, 13, 14, 15);
            v16h bfrag = *reinterpret_cast<const v16h*>(bcol + j0 + off + kb);
            acc = __builtin_amdgcn_wmma_f32_16x16x32_f16(
                      false, afrag, false, bfrag, (short)0, acc, false, false);
        }
    }

    int rb = (lane < 16) ? 0 : 8;
    union { v8f v; float f[8]; } U; U.v = acc;
#pragma unroll
    for (int r = 0; r < 8; ++r) {
        hp[((size_t)bh * NNODE + i0 + rb + r) * HID + col] = U.f[r];
    }
}

// ---------------------------------------------------------------------------
// sf_out = elu(mean_h hp) + residual
// ---------------------------------------------------------------------------
__global__ void combine_kernel(const float* __restrict__ hp,
                               const float* __restrict__ res,
                               float* __restrict__ out) {
    int idx = blockIdx.x * blockDim.x + threadIdx.x;   // over B*N*HID
    if (idx >= BATCH * NNODE * HID) return;
    int f = idx % HID; int bn = idx / HID;
    int b = bn / NNODE, n = bn % NNODE;
    float a = 0.f;
    for (int h = 0; h < NHEAD; ++h)
        a += hp[((size_t)(b * NHEAD + h) * NNODE + n) * HID + f];
    a *= (1.0f / 3.0f);
    a = a > 0.f ? a : (__expf(a) - 1.0f);               // elu
    if (res) a += res[idx];
    out[idx] = a;
}

// ---------------------------------------------------------------------------
// f32 -> f16 elementwise convert
// ---------------------------------------------------------------------------
__global__ void conv_kernel(const float* __restrict__ in,
                            half_t* __restrict__ out, int n) {
    int idx = blockIdx.x * blockDim.x + threadIdx.x;
    if (idx < n) out[idx] = (half_t)in[idx];
}

// ---------------------------------------------------------------------------
// GRU input GEMM via WMMA:  C[GM x 192] = A[GM x 64] * W^T + bias
// W given as torch (192,64) row-major == exactly the contiguous-K B layout.
// Block = 4 waves = 16 rows x 64 cols; all fragments direct global loads.
// ---------------------------------------------------------------------------
__global__ void __launch_bounds__(128) gru_gemm_kernel(
    const half_t* __restrict__ A, const half_t* __restrict__ BT,
    const float* __restrict__ bias, float* __restrict__ C) {
    const int rtiles = GM / 16;          // 512
    int rtile = blockIdx.x % rtiles;
    int ctile = blockIdx.x / rtiles;     // 0..2
    int tid = threadIdx.x, wave = tid >> 5, lane = tid & 31;
    int row = rtile * 16 + (lane & 15);
    int col = ctile * 64 + wave * 16 + (lane & 15);
    int klo = (lane < 16) ? 0 : 16;
    int kb  = (lane < 16) ? 0 : 16;

    v8f acc = {0.f, 0.f, 0.f, 0.f, 0.f, 0.f, 0.f, 0.f};
#pragma unroll
    for (int c = 0; c < 2; ++c) {
        int off = c * 32;
        v8h alo = *reinterpret_cast<const v8h*>(A + (size_t)row * 64 + off + klo);
        v8h ahi = *reinterpret_cast<const v8h*>(A + (size_t)row * 64 + off + klo + 8);
        v16h afrag = __builtin_shufflevector(alo, ahi,
                       0, 1, 2, 3, 4, 5, 6, 7, 8, 9, 10, 11, 12, 13, 14, 15);
        v16h bfrag = *reinterpret_cast<const v16h*>(BT + (size_t)col * 64 + off + kb);
        acc = __builtin_amdgcn_wmma_f32_16x16x32_f16(
                  false, afrag, false, bfrag, (short)0, acc, false, false);
    }
    float bv = bias[col];
    int rb = (lane < 16) ? 0 : 8;
    union { v8f v; float f[8]; } U; U.v = acc;
#pragma unroll
    for (int r = 0; r < 8; ++r) {
        C[(size_t)(rtile * 16 + rb + r) * FDIM + col] = U.f[r] + bv;
    }
}

// ---------------------------------------------------------------------------
// GRU gates (torch semantics), elementwise over (b,n,k).
// ---------------------------------------------------------------------------
__global__ void gates_kernel(const float* __restrict__ gi,
                             const float* __restrict__ gh,
                             const float* __restrict__ hprev,
                             float* __restrict__ hnew) {
    int idx = blockIdx.x * blockDim.x + threadIdx.x;   // over B*N*HID
    if (idx >= BATCH * NNODE * HID) return;
    int bn = idx / HID, k = idx % HID;
    size_t g = (size_t)bn * FDIM;
    float ir = gi[g + k], iz = gi[g + HID + k], in = gi[g + 2 * HID + k];
    float hr = gh[g + k], hz = gh[g + HID + k], hn = gh[g + 2 * HID + k];
    float r = 1.f / (1.f + __expf(-(ir + hr)));
    float z = 1.f / (1.f + __expf(-(iz + hz)));
    float n = tanhf(in + r * hn);
    hnew[idx] = (1.f - z) * n + z * hprev[idx];
}

// ---------------------------------------------------------------------------
// Actor (64->32->2 + softmax) and critic (64->32->1). One block per (b,n).
// ---------------------------------------------------------------------------
__global__ void heads_kernel(const float* __restrict__ hnew,
                             const float* __restrict__ Wa1, const float* __restrict__ ba1,
                             const float* __restrict__ Wa2, const float* __restrict__ ba2,
                             const float* __restrict__ Wc1, const float* __restrict__ bc1,
                             const float* __restrict__ Wc2, const float* __restrict__ bc2,
                             float* __restrict__ probs, float* __restrict__ value) {
    int bn = blockIdx.x;
    int t  = threadIdx.x;                // 0..63
    __shared__ float hrow[HID];
    __shared__ float a1[32], c1[32];
    hrow[t] = hnew[(size_t)bn * HID + t];
    __syncthreads();
    if (t < 32) {
        float a = ba1[t];
        for (int f = 0; f < HID; ++f) a += hrow[f] * Wa1[f * 32 + t];
        a1[t] = a > 0.f ? a : 0.f;
    } else {
        int u = t - 32;
        float c = bc1[u];
        for (int f = 0; f < HID; ++f) c += hrow[f] * Wc1[f * 32 + u];
        c1[u] = c > 0.f ? c : 0.f;
    }
    __syncthreads();
    if (t == 0) {
        float l0 = ba2[0], l1 = ba2[1];
        for (int i = 0; i < 32; ++i) { l0 += a1[i] * Wa2[i * 2]; l1 += a1[i] * Wa2[i * 2 + 1]; }
        float m = fmaxf(l0, l1);
        float e0 = __expf(l0 - m), e1 = __expf(l1 - m);
        float inv = 1.f / (e0 + e1);
        probs[(size_t)bn * 2]     = e0 * inv;
        probs[(size_t)bn * 2 + 1] = e1 * inv;
    } else if (t == 32) {
        float v = bc2[0];
        for (int i = 0; i < 32; ++i) v += c1[i] * Wc2[i];
        value[bn] = v;
    }
}

// ---------------------------------------------------------------------------
extern "C" void kernel_launch(void* const* d_in, const int* in_sizes, int n_in,
                              void* d_out, int out_size, void* d_ws, size_t ws_size,
                              hipStream_t stream) {
    (void)in_sizes; (void)n_in; (void)out_size; (void)ws_size;
    const float* x      = (const float*)d_in[0];
    const int*   adj    = (const int*)d_in[1];
    const float* h_prev = (const float*)d_in[2];
    const float* W[3]    = {(const float*)d_in[3], (const float*)d_in[6], (const float*)d_in[9]};
    const float* asrc[3] = {(const float*)d_in[4], (const float*)d_in[7], (const float*)d_in[10]};
    const float* adst[3] = {(const float*)d_in[5], (const float*)d_in[8], (const float*)d_in[11]};
    const float* gwi = (const float*)d_in[12];
    const float* gwh = (const float*)d_in[13];
    const float* gbi = (const float*)d_in[14];
    const float* gbh = (const float*)d_in[15];
    const float* Wa1 = (const float*)d_in[16]; const float* ba1 = (const float*)d_in[17];
    const float* Wa2 = (const float*)d_in[18]; const float* ba2 = (const float*)d_in[19];
    const float* Wc1 = (const float*)d_in[20]; const float* bc1 = (const float*)d_in[21];
    const float* Wc2 = (const float*)d_in[22]; const float* bc2 = (const float*)d_in[23];

    const size_t BHNF = (size_t)BATCH * NHEAD * NNODE * HID;   // 1.57M
    const size_t BHN  = (size_t)BATCH * NHEAD * NNODE;         // 24576
    const size_t BNF  = (size_t)BATCH * NNODE * HID;           // 524288

    char* ws = (char*)d_ws;
    size_t off = 0;
    auto alloc = [&](size_t bytes) -> char* {
        char* p = ws + off;
        off += (bytes + 255) & ~(size_t)255;
        return p;
    };
    float*  wh32  = (float*) alloc(BHNF * 4);
    half_t* whT16 = (half_t*)alloc(BHNF * 2);
    float*  hp    = (float*) alloc(BHNF * 4);
    float*  sb    = (float*) alloc(BHN * 4);
    float*  db    = (float*) alloc(BHN * 4);
    float*  mb    = (float*) alloc(BHN * 4);
    float*  ib    = (float*) alloc(BHN * 4);
    float*  sf0   = (float*) alloc(BNF * 4);
    float*  sf1   = (float*) alloc(BNF * 4);
    float*  sf2   = (float*) alloc(BNF * 4);
    half_t* sf216 = (half_t*)alloc(BNF * 2);
    half_t* hpr16 = (half_t*)alloc(BNF * 2);
    half_t* wi16  = (half_t*)alloc((size_t)FDIM * HID * 2);
    half_t* whw16 = (half_t*)alloc((size_t)FDIM * HID * 2);
    float*  gi    = (float*) alloc((size_t)GM * FDIM * 4);
    float*  gh    = (float*) alloc((size_t)GM * FDIM * 4);

    float* out   = (float*)d_out;
    float* probs = out;
    float* value = out + (size_t)BATCH * NNODE * 2;
    float* hnew  = out + (size_t)BATCH * NNODE * 3;

    float* outs[3] = {sf0, sf1, sf2};
    const float* cur = x;
    const float* res = nullptr;
    int fin = 3;
    for (int l = 0; l < 3; ++l) {
        wh_kernel<<<BATCH * NNODE, FDIM, 0, stream>>>(
            cur, fin, W[l], asrc[l], adst[l], wh32, sb, db);
        transpose_kernel<<<BATCH * NHEAD * (NNODE / 64), 256, 0, stream>>>(
            wh32, whT16);
        stats_kernel<<<(BATCH * NHEAD * NNODE) / 8, 256, 0, stream>>>(
            sb, db, adj, mb, ib);
        attn_kernel<<<BATCH * NHEAD * (NNODE / 16), 128, 0, stream>>>(
            whT16, sb, db, mb, ib, adj, hp);
        combine_kernel<<<(BATCH * NNODE * HID + 255) / 256, 256, 0, stream>>>(
            hp, res, outs[l]);
        res = outs[l];
        cur = outs[l];
        fin = HID;
    }
    // GRU: f16 converts + two WMMA GEMMs + elementwise gates
    conv_kernel<<<((int)BNF + 255) / 256, 256, 0, stream>>>(sf2, sf216, (int)BNF);
    conv_kernel<<<((int)BNF + 255) / 256, 256, 0, stream>>>(h_prev, hpr16, (int)BNF);
    conv_kernel<<<(FDIM * HID + 255) / 256, 256, 0, stream>>>(gwi, wi16, FDIM * HID);
    conv_kernel<<<(FDIM * HID + 255) / 256, 256, 0, stream>>>(gwh, whw16, FDIM * HID);
    gru_gemm_kernel<<<(GM / 16) * 3, 128, 0, stream>>>(sf216, wi16, gbi, gi);
    gru_gemm_kernel<<<(GM / 16) * 3, 128, 0, stream>>>(hpr16, whw16, gbh, gh);
    gates_kernel<<<((int)BNF + 255) / 256, 256, 0, stream>>>(gi, gh, h_prev, hnew);
    heads_kernel<<<BATCH * NNODE, 64, 0, stream>>>(
        hnew, Wa1, ba1, Wa2, ba2, Wc1, bc1, Wc2, bc2, probs, value);
}
